// LinformerAttention_53145925320735
// MI455X (gfx1250) — matrix operand order repaired and checked by reading, not compiled, not run on hardware
//
#include <hip/hip_runtime.h>

typedef __attribute__((ext_vector_type(16))) __bf16 v16bf;
typedef __attribute__((ext_vector_type(8)))  float  v8f;
typedef unsigned short u16;
typedef unsigned int   u32;

#define HIDDEN  1024
#define HEADS   16
#define HEADDIM 64
#define KPROJ   256
#define SEQ     4096
#define BSZ     4
#define ROWS    (BSZ * SEQ)   // 16384

// ---------------------------------------------------------------- helpers

static __device__ __forceinline__ u16 to_bf16(float f) {
  __bf16 b = (__bf16)f;                       // hw convert (v_cvt_*bf16*)
  return __builtin_bit_cast(u16, b);
}

static __device__ __forceinline__ u32 to_bf16x2(float lo, float hi) {
  return (u32)to_bf16(lo) | ((u32)to_bf16(hi) << 16);
}

static __device__ __forceinline__ v16bf make_v16bf(uint4 lo, uint4 hi) {
  union { uint4 u[2]; v16bf v; } c;
  c.u[0] = lo; c.u[1] = hi;
  return c.v;
}

// A fragment 16x32 (MxK), bf16 row-major, K contiguous. `A` points at (m0,k0).
// ISA layout: lane L holds row m=L&15; halves 0..7 = K (L>>4)*8+0..7,
// halves 8..15 = K 16+(L>>4)*8+0..7.
static __device__ __forceinline__ v16bf load_a_frag(const u16* A, int lda, int lane) {
  const int m = lane & 15, h = lane >> 4;
  const u16* p = A + m * lda + h * 8;
  return make_v16bf(*(const uint4*)p, *(const uint4*)(p + 16));
}

// B fragment 32x16 (KxN) for NT gemm: B stored as (N,K) row-major, K contiguous.
// ISA layout: lane L holds col n=L&15; halves 0..7 = K (L>>4)*8+0..7,
// halves 8..15 = K 16+(L>>4)*8+0..7.
static __device__ __forceinline__ v16bf load_b_nt(const u16* B, int ldb, int lane) {
  const int n = lane & 15, h = lane >> 4;
  const u16* p = B + n * ldb + h * 8;
  return make_v16bf(*(const uint4*)p, *(const uint4*)(p + 16));
}

static __device__ __forceinline__ v8f wmma_bf16(v16bf a, v16bf b, v8f c) {
  return __builtin_amdgcn_wmma_f32_16x16x32_bf16(false, a, false, b, (short)0, c,
                                                 false, false);
}

static __device__ __forceinline__ v8f v8f_zero() {
  v8f z = {0.f, 0.f, 0.f, 0.f, 0.f, 0.f, 0.f, 0.f};
  return z;
}

// ---------------------------------------------------------------- conversion

// n2 = element count / 2; packs two bf16 per 32-bit store.
__global__ void f32_to_bf16_kernel(const float2* __restrict__ in,
                                   u32* __restrict__ out, int n2) {
  int i = blockIdx.x * blockDim.x + threadIdx.x;
  if (i < n2) {
    float2 v = in[i];
    out[i] = to_bf16x2(v.x, v.y);
  }
}

// out[c*rows + r] = bf16(in[r*cols + c])   (E/F -> E^T/F^T, bf16)
__global__ void transpose_f32_to_bf16_kernel(const float* __restrict__ in,
                                             u16* __restrict__ out,
                                             int rows, int cols) {
  int i = blockIdx.x * blockDim.x + threadIdx.x;
  if (i < rows * cols) {
    int r = i / cols, c = i % cols;
    out[c * rows + r] = to_bf16(in[i]);
  }
}

// ---------------------------------------------------------------- NT GEMM
// C(M,N) = A(M,K) * B(N,K)^T, bf16 in, f32 accumulate, one wave per 64x64 tile.
// store_mode: 0 = f32 row-major, 1 = bf16 row-major,
//             2 = bf16 per-head transposed: out[((b*HEADS+h)*HEADDIM+d)*SEQ + s]
//                 where row = b*SEQ+s, col = h*HEADDIM+d  (for K/V projections)
__global__ __launch_bounds__(32) void gemm_nt_bf16_kernel(
    const u16* __restrict__ A, const u16* __restrict__ B, void* __restrict__ Cout,
    int K, int lda, int ldb, int ldc, int store_mode) {
  const int lane = threadIdx.x;
  const int n0 = blockIdx.x * 64;
  const int m0 = blockIdx.y * 64;

  v8f acc[4][4];
#pragma unroll
  for (int i = 0; i < 4; ++i)
#pragma unroll
    for (int j = 0; j < 4; ++j) acc[i][j] = v8f_zero();

  for (int k0 = 0; k0 < K; k0 += 32) {
    v16bf af[4], bf[4];
#pragma unroll
    for (int i = 0; i < 4; ++i)
      af[i] = load_a_frag(A + (m0 + 16 * i) * lda + k0, lda, lane);
#pragma unroll
    for (int j = 0; j < 4; ++j)
      bf[j] = load_b_nt(B + (n0 + 16 * j) * ldb + k0, ldb, lane);
#pragma unroll
    for (int i = 0; i < 4; ++i)
#pragma unroll
      for (int j = 0; j < 4; ++j) acc[i][j] = wmma_bf16(af[i], bf[j], acc[i][j]);
  }

  const int mr = (lane >> 4) * 8, nc = lane & 15;
  if (store_mode == 0) {
    float* C = (float*)Cout;
#pragma unroll
    for (int i = 0; i < 4; ++i)
#pragma unroll
      for (int j = 0; j < 4; ++j)
#pragma unroll
        for (int r = 0; r < 8; ++r)
          C[(m0 + 16 * i + mr + r) * ldc + n0 + 16 * j + nc] = acc[i][j][r];
  } else if (store_mode == 1) {
    u16* C = (u16*)Cout;
#pragma unroll
    for (int i = 0; i < 4; ++i)
#pragma unroll
      for (int j = 0; j < 4; ++j)
#pragma unroll
        for (int r = 0; r < 8; ++r)
          C[(m0 + 16 * i + mr + r) * ldc + n0 + 16 * j + nc] =
              to_bf16(acc[i][j][r]);
  } else {
    // per-head transposed store: (b, h, d, s) with s contiguous
    u16* C = (u16*)Cout;
#pragma unroll
    for (int i = 0; i < 4; ++i)
#pragma unroll
      for (int j = 0; j < 4; ++j)
#pragma unroll
        for (int r = 0; r < 8; ++r) {
          int row = m0 + 16 * i + mr + r;       // b*SEQ + s
          int col = n0 + 16 * j + nc;           // h*HEADDIM + d
          int b = row >> 12, s = row & (SEQ - 1);
          C[(((b << 4) | (col >> 6)) * HEADDIM + (col & 63)) * SEQ + s] =
              to_bf16(acc[i][j][r]);
        }
  }
}

// ---------------------------------------------------------------- KV projection
// Out[bh](256,64) = Et(256,4096) @ head_slice^T(64,4096)^T  -- all NT now:
//   A = Et (p, s) K-contiguous, B = KVt[bh] (d, s) K-contiguous.
// transpose_out=1 stores Out as (64,256) so the later PV gemm is NT.
__global__ __launch_bounds__(32) void kvproj_kernel(
    const u16* __restrict__ Et, const u16* __restrict__ KVt,
    u16* __restrict__ Out, int transpose_out) {
  const int lane = threadIdx.x;
  const int m0 = blockIdx.x * 64;              // projection-dim offset
  const int bh = blockIdx.y;
  const u16* Bbase = KVt + bh * HEADDIM * SEQ; // (64, 4096), s contiguous

  v8f acc[4][4];
#pragma unroll
  for (int i = 0; i < 4; ++i)
#pragma unroll
    for (int j = 0; j < 4; ++j) acc[i][j] = v8f_zero();

  for (int k0 = 0; k0 < SEQ; k0 += 32) {
    v16bf af[4], bf[4];
#pragma unroll
    for (int i = 0; i < 4; ++i)
      af[i] = load_a_frag(Et + (m0 + 16 * i) * SEQ + k0, SEQ, lane);
#pragma unroll
    for (int j = 0; j < 4; ++j)
      bf[j] = load_b_nt(Bbase + (16 * j) * SEQ + k0, SEQ, lane);
#pragma unroll
    for (int i = 0; i < 4; ++i)
#pragma unroll
      for (int j = 0; j < 4; ++j) acc[i][j] = wmma_bf16(af[i], bf[j], acc[i][j]);
  }

  const int mr = (lane >> 4) * 8, nc = lane & 15;
  if (!transpose_out) {
    u16* O = Out + bh * KPROJ * HEADDIM;       // (256, 64) row-major
#pragma unroll
    for (int i = 0; i < 4; ++i)
#pragma unroll
      for (int j = 0; j < 4; ++j)
#pragma unroll
        for (int r = 0; r < 8; ++r)
          O[(m0 + 16 * i + mr + r) * HEADDIM + 16 * j + nc] =
              to_bf16(acc[i][j][r]);
  } else {
    u16* O = Out + bh * HEADDIM * KPROJ;       // (64, 256) row-major
#pragma unroll
    for (int i = 0; i < 4; ++i)
#pragma unroll
      for (int j = 0; j < 4; ++j)
#pragma unroll
        for (int r = 0; r < 8; ++r)
          O[(16 * j + nc) * KPROJ + m0 + 16 * i + mr + r] =
              to_bf16(acc[i][j][r]);
  }
}

// ---------------------------------------------------------------- fused attention
// One wave per (b*h, 16 query rows): scores (16x256) -> softmax -> attn @ VpT.
__global__ __launch_bounds__(32) void attention_kernel(
    const u16* __restrict__ Q,      // (16384, 1024) bf16
    const u16* __restrict__ Kp,     // (64, 256, 64)  bf16
    const u16* __restrict__ VpT,    // (64, 64, 256)  bf16
    u16* __restrict__ AO) {         // (16384, 1024) bf16
  __shared__ __align__(16) u16 attn_s[16 * KPROJ];

  const int lane = threadIdx.x;
  const int q0 = blockIdx.x * 16;
  const int bh = blockIdx.y;
  const int b = bh >> 4, h = bh & 15;

  const u16* Qbase = Q + (b * SEQ + q0) * HIDDEN + h * HEADDIM;
  const u16* Kbase = Kp + bh * KPROJ * HEADDIM;

  // scores: 16 tiles of 16x16 over the 256 projection columns, K = 64.
  v8f sc[16];
#pragma unroll
  for (int t = 0; t < 16; ++t) sc[t] = v8f_zero();
#pragma unroll
  for (int ks = 0; ks < 2; ++ks) {
    v16bf aq = load_a_frag(Qbase + ks * 32, HIDDEN, lane);
#pragma unroll
    for (int t = 0; t < 16; ++t) {
      v16bf bk = load_b_nt(Kbase + t * 16 * HEADDIM + ks * 32, HEADDIM, lane);
      sc[t] = wmma_bf16(aq, bk, sc[t]);
    }
  }

  // softmax over the 256 columns. Row m = (lane>>4)*8 + r lives in the 16
  // lanes sharing lane>>4; xor-shuffles with masks 1,2,4,8 stay in-group.
  float rmax[8], rsum[8];
#pragma unroll
  for (int r = 0; r < 8; ++r) rmax[r] = -3.402823466e38f;
#pragma unroll
  for (int t = 0; t < 16; ++t)
#pragma unroll
    for (int r = 0; r < 8; ++r) {
      sc[t][r] *= 0.125f;                       // 1/sqrt(64)
      rmax[r] = fmaxf(rmax[r], sc[t][r]);
    }
#pragma unroll
  for (int m = 1; m <= 8; m <<= 1)
#pragma unroll
    for (int r = 0; r < 8; ++r)
      rmax[r] = fmaxf(rmax[r], __shfl_xor(rmax[r], m, 32));
#pragma unroll
  for (int r = 0; r < 8; ++r) rsum[r] = 0.f;
#pragma unroll
  for (int t = 0; t < 16; ++t)
#pragma unroll
    for (int r = 0; r < 8; ++r) {
      float e = __expf(sc[t][r] - rmax[r]);
      sc[t][r] = e;
      rsum[r] += e;
    }
#pragma unroll
  for (int m = 1; m <= 8; m <<= 1)
#pragma unroll
    for (int r = 0; r < 8; ++r) rsum[r] += __shfl_xor(rsum[r], m, 32);

  const int mr = (lane >> 4) * 8, nc = lane & 15;
#pragma unroll
  for (int r = 0; r < 8; ++r) {
    float inv = 1.0f / rsum[r];
#pragma unroll
    for (int t = 0; t < 16; ++t)
      attn_s[(mr + r) * KPROJ + t * 16 + nc] = to_bf16(sc[t][r] * inv);
  }
  __syncthreads();   // single-wave WG: orders DS stores before DS loads

  // out = attn(16x256) @ Vp(256x64); B = VpT (64,256) -> NT, contiguous.
  const u16* Vbase = VpT + bh * HEADDIM * KPROJ;
  v8f oacc[4];
#pragma unroll
  for (int j = 0; j < 4; ++j) oacc[j] = v8f_zero();
#pragma unroll
  for (int ks = 0; ks < 8; ++ks) {
    const int m = lane & 15, hh = lane >> 4;
    const u16* p = &attn_s[m * KPROJ + ks * 32 + hh * 8];
    v16bf aa = make_v16bf(*(const uint4*)p, *(const uint4*)(p + 16));
#pragma unroll
    for (int j = 0; j < 4; ++j) {
      v16bf bv = load_b_nt(Vbase + j * 16 * KPROJ + ks * 32, KPROJ, lane);
      oacc[j] = wmma_bf16(aa, bv, oacc[j]);
    }
  }

  u16* Obase = AO + (b * SEQ + q0) * HIDDEN + h * HEADDIM;
#pragma unroll
  for (int j = 0; j < 4; ++j)
#pragma unroll
    for (int r = 0; r < 8; ++r)
      Obase[(mr + r) * HIDDEN + j * 16 + nc] = to_bf16(oacc[j][r]);
}

// ---------------------------------------------------------------- launch

extern "C" void kernel_launch(void* const* d_in, const int* in_sizes, int n_in,
                              void* d_out, int out_size, void* d_ws, size_t ws_size,
                              hipStream_t stream) {
  (void)in_sizes; (void)n_in; (void)out_size; (void)ws_size;
  const float* h  = (const float*)d_in[0];
  const float* Wq = (const float*)d_in[1];
  const float* Wk = (const float*)d_in[2];
  const float* Wv = (const float*)d_in[3];
  const float* Wo = (const float*)d_in[4];
  const float* E  = (const float*)d_in[5];
  const float* F  = (const float*)d_in[6];

  char* ws = (char*)d_ws;
  size_t off = 0;
  auto alloc = [&](size_t bytes) -> char* {
    char* p = ws + off;
    off += (bytes + 255) & ~(size_t)255;
    return p;
  };
  const size_t big = (size_t)ROWS * HIDDEN * sizeof(u16);       // 33.5 MB
  const size_t wsz = (size_t)HIDDEN * HIDDEN * sizeof(u16);     // 2 MB
  const size_t psz = (size_t)KPROJ * SEQ * sizeof(u16);         // 2 MB

  u16* X   = (u16*)alloc(big);     // bf16 hidden states; reused as attn-out
  u16* Qb  = (u16*)alloc(big);     // (16384, 1024) row-major
  u16* Kt  = (u16*)alloc(big);     // (b, h, d, s) per-head transposed
  u16* Vt  = (u16*)alloc(big);     // (b, h, d, s) per-head transposed
  u16* Wqb = (u16*)alloc(wsz);
  u16* Wkb = (u16*)alloc(wsz);
  u16* Wvb = (u16*)alloc(wsz);
  u16* Wob = (u16*)alloc(wsz);
  u16* Etb = (u16*)alloc(psz);     // (256, 4096)
  u16* Ftb = (u16*)alloc(psz);     // (256, 4096)
  u16* Kp  = (u16*)alloc((size_t)64 * KPROJ * HEADDIM * sizeof(u16)); // (bh,256,64)
  u16* VpT = (u16*)alloc((size_t)64 * HEADDIM * KPROJ * sizeof(u16)); // (bh,64,256)

  // 1) convert to bf16 (packed float2 -> u32)
  {
    int n2 = (ROWS * HIDDEN) / 2;
    f32_to_bf16_kernel<<<(n2 + 255) / 256, 256, 0, stream>>>((const float2*)h, (u32*)X, n2);
    int nw2 = (HIDDEN * HIDDEN) / 2;
    f32_to_bf16_kernel<<<(nw2 + 255) / 256, 256, 0, stream>>>((const float2*)Wq, (u32*)Wqb, nw2);
    f32_to_bf16_kernel<<<(nw2 + 255) / 256, 256, 0, stream>>>((const float2*)Wk, (u32*)Wkb, nw2);
    f32_to_bf16_kernel<<<(nw2 + 255) / 256, 256, 0, stream>>>((const float2*)Wv, (u32*)Wvb, nw2);
    f32_to_bf16_kernel<<<(nw2 + 255) / 256, 256, 0, stream>>>((const float2*)Wo, (u32*)Wob, nw2);
    int ne = SEQ * KPROJ;
    transpose_f32_to_bf16_kernel<<<(ne + 255) / 256, 256, 0, stream>>>(E, Etb, SEQ, KPROJ);
    transpose_f32_to_bf16_kernel<<<(ne + 255) / 256, 256, 0, stream>>>(F, Ftb, SEQ, KPROJ);
  }

  // 2) Q/K/V projections: (16384,1024) @ W^T ; K,V stored per-head transposed
  dim3 gproj(HIDDEN / 64, ROWS / 64);   // (16, 256)
  gemm_nt_bf16_kernel<<<gproj, 32, 0, stream>>>(X, Wqb, Qb, HIDDEN, HIDDEN, HIDDEN, HIDDEN, 1);
  gemm_nt_bf16_kernel<<<gproj, 32, 0, stream>>>(X, Wkb, Kt, HIDDEN, HIDDEN, HIDDEN, 0, 2);
  gemm_nt_bf16_kernel<<<gproj, 32, 0, stream>>>(X, Wvb, Vt, HIDDEN, HIDDEN, HIDDEN, 0, 2);

  // 3) Linformer K/V projection per (b,h) -- pure NT GEMMs now
  kvproj_kernel<<<dim3(KPROJ / 64, 64), 32, 0, stream>>>(Etb, Kt, Kp, 0);
  kvproj_kernel<<<dim3(KPROJ / 64, 64), 32, 0, stream>>>(Ftb, Vt, VpT, 1);

  // 4) fused scores + softmax + PV; writes attn-out into X (X is dead now)
  attention_kernel<<<dim3(SEQ / 16, 64), 32, 0, stream>>>(Qb, Kp, VpT, X);

  // 5) output projection -> f32 d_out
  gemm_nt_bf16_kernel<<<gproj, 32, 0, stream>>>(X, Wob, d_out, HIDDEN, HIDDEN, HIDDEN, HIDDEN, 0);
}
